// GAT0mp_30107720745832
// MI455X (gfx1250) — matrix operand order repaired and checked by reading, not compiled, not run on hardware
//
#include <hip/hip_runtime.h>

// ---------------------------------------------------------------------------
// GATv2 x3 + mean-pool + MLP head for MI455X (gfx1250, wave32, WMMA).
// Dense projections: v_wmma_f32_16x16x32_bf16, W pre-packed into B-fragment
// order so all WMMA operands load with global_load_b128.  Each wave computes
// a 16x64 output strip (4 WMMAs per A fragment).  Edge attention is
// wave-per-edge with We staged in LDS; softmax via atomic segment max
// (monotone uint mapping) + atomic segment sum.
// ---------------------------------------------------------------------------

#define NNODES 16384
#define NEDGES 131072
#define NETOT  (NNODES + NEDGES)
#define EDIM   32
#define NGRAPH 128
#define NEG_SLOPE 0.2f
#define BN_EPS 1e-5f

typedef __attribute__((ext_vector_type(16))) __bf16 v16bf;
typedef __attribute__((ext_vector_type(8)))  float  v8f;

// ---- float <-> monotone uint mapping for atomic float max -----------------
__device__ __forceinline__ unsigned mapf(float f) {
  unsigned u = __float_as_uint(f);
  return (u & 0x80000000u) ? ~u : (u | 0x80000000u);
}
__device__ __forceinline__ float unmapf(unsigned m) {
  unsigned u = (m & 0x80000000u) ? (m ^ 0x80000000u) : ~m;
  return __uint_as_float(u);
}

// ---- elementwise fp32 -> bf16 (activations, row-major kept) ---------------
__global__ void k_f2bf(const float* __restrict__ in, __bf16* __restrict__ out, int n) {
  int i = blockIdx.x * blockDim.x + threadIdx.x;
  if (i < n) out[i] = (__bf16)in[i];
}

// ---- W[K,Nc] fp32 -> packed bf16 B-fragments ------------------------------
// Packed element [( (nt*(K/32) + kt)*32 + lane )*16 + i] =
//     W[kt*32 + 16*(lane>>4) + i][nt*16 + (lane&15)]
// so each lane's 16 K-consecutive values are contiguous (2 x b128 per frag).
__global__ void k_repackW(const float* __restrict__ W, __bf16* __restrict__ out,
                          int K, int Nc) {
  int t = blockIdx.x * blockDim.x + threadIdx.x;
  int kt32 = K >> 5;
  int total = (Nc >> 4) * kt32 * 32;
  if (t >= total) return;
  int lane = t & 31;
  int kt = (t >> 5) % kt32;
  int nt = (t >> 5) / kt32;
  int half = lane >> 4, col = lane & 15;
  const float* src = W + (size_t)(kt * 32 + half * 16) * Nc + nt * 16 + col;
  __bf16* dst = out + (size_t)t * 16;
#pragma unroll
  for (int i = 0; i < 16; ++i) dst[i] = (__bf16)src[(size_t)i * Nc];
}

// ---- src/dst concat (edges then self loops) -------------------------------
__global__ void k_fill_cat(const int* __restrict__ ei, int* __restrict__ srcc,
                           int* __restrict__ dstc) {
  int i = blockIdx.x * blockDim.x + threadIdx.x;
  if (i >= NETOT) return;
  if (i < NEDGES) { srcc[i] = ei[i]; dstc[i] = ei[NEDGES + i]; }
  else            { srcc[i] = i - NEDGES; dstc[i] = i - NEDGES; }
}

__global__ void k_deg(const int* __restrict__ ei, float* __restrict__ deg) {
  int i = blockIdx.x * blockDim.x + threadIdx.x;
  if (i < NEDGES) atomicAdd(&deg[ei[NEDGES + i]], 1.0f);
}

__global__ void k_loop_acc(const int* __restrict__ ei, const float* __restrict__ eattr,
                           float* __restrict__ ea_cat) {
  int t = blockIdx.x * blockDim.x + threadIdx.x;
  if (t >= NEDGES * EDIM) return;
  int e = t >> 5, j = t & 31;
  int d = ei[NEDGES + e];
  atomicAdd(&ea_cat[((size_t)(NEDGES + d)) * EDIM + j], eattr[(size_t)e * EDIM + j]);
}

__global__ void k_loop_div(const float* __restrict__ deg, float* __restrict__ ea_cat) {
  int t = blockIdx.x * blockDim.x + threadIdx.x;
  if (t >= NNODES * EDIM) return;
  int n = t >> 5, j = t & 31;
  ea_cat[((size_t)(NEDGES + n)) * EDIM + j] /= fmaxf(deg[n], 1.0f);
}

// ---- WMMA GEMM: C[M,Nc] = A[M,K](bf16) @ W(packed bf16) + bias ------------
// One wave -> 16x64 strip (4 N-tiles), A fragment reused across 4 WMMAs.
__global__ __launch_bounds__(256)
void k_gemm_bf16(const __bf16* __restrict__ A, const __bf16* __restrict__ Bpk,
                 const float* __restrict__ bias, float* __restrict__ C,
                 int M, int K, int Nc) {
  int wave = (blockIdx.x * blockDim.x + threadIdx.x) >> 5;
  int lane = threadIdx.x & 31;
  int ngrp = Nc >> 6;                       // groups of 4 n-tiles
  int mt = wave / ngrp, ng = wave % ngrp;
  if (mt * 16 >= M) return;                 // wave-uniform exit
  int row = lane & 15, half = lane >> 4;
  int kt32 = K >> 5;

  const __bf16* ap = A + (size_t)(mt * 16 + row) * K + half * 16;
  const __bf16* bp = Bpk + (((size_t)(ng * 4) * kt32) * 32 + lane) * 16;
  const size_t bstep = (size_t)32 * 16;     // one K-step within an n-tile
  const size_t bnt   = (size_t)kt32 * bstep;

  v8f acc0 = {}, acc1 = {}, acc2 = {}, acc3 = {};
  for (int kt = 0; kt < kt32; ++kt) {
    union { v16bf v; uint4 q[2]; } a, b0, b1, b2, b3;
    a.q[0] = *reinterpret_cast<const uint4*>(ap);
    a.q[1] = *reinterpret_cast<const uint4*>(ap + 8);
    const __bf16* bb = bp + (size_t)kt * bstep;
    b0.q[0] = *reinterpret_cast<const uint4*>(bb);
    b0.q[1] = *reinterpret_cast<const uint4*>(bb + 8);
    b1.q[0] = *reinterpret_cast<const uint4*>(bb + bnt);
    b1.q[1] = *reinterpret_cast<const uint4*>(bb + bnt + 8);
    b2.q[0] = *reinterpret_cast<const uint4*>(bb + 2 * bnt);
    b2.q[1] = *reinterpret_cast<const uint4*>(bb + 2 * bnt + 8);
    b3.q[0] = *reinterpret_cast<const uint4*>(bb + 3 * bnt);
    b3.q[1] = *reinterpret_cast<const uint4*>(bb + 3 * bnt + 8);
    __builtin_prefetch((const void*)(ap + 32), 0, 0);        // next A K-step
    __builtin_prefetch((const void*)(bb + bstep), 0, 0);     // next B K-step

    acc0 = __builtin_amdgcn_wmma_f32_16x16x32_bf16(false, a.v, false, b0.v,
                                                   (short)0, acc0, false, false);
    acc1 = __builtin_amdgcn_wmma_f32_16x16x32_bf16(false, a.v, false, b1.v,
                                                   (short)0, acc1, false, false);
    acc2 = __builtin_amdgcn_wmma_f32_16x16x32_bf16(false, a.v, false, b2.v,
                                                   (short)0, acc2, false, false);
    acc3 = __builtin_amdgcn_wmma_f32_16x16x32_bf16(false, a.v, false, b3.v,
                                                   (short)0, acc3, false, false);
    ap += 32;
  }

#pragma unroll
  for (int t = 0; t < 4; ++t) {
    v8f acc = t == 0 ? acc0 : t == 1 ? acc1 : t == 2 ? acc2 : acc3;
    int colbase = (ng * 4 + t) * 16;
    float bv = bias[colbase + row];
#pragma unroll
    for (int i = 0; i < 8; ++i) {
      int r = mt * 16 + i + 8 * half;
      C[(size_t)r * Nc + colbase + row] = acc[i] + bv;
    }
  }
}

// ---- attention logits: wave per edge, We staged in LDS --------------------
// e = att . leaky_relu(xl[src] + xr[dst] + ea@We);  atomic segment max.
__global__ __launch_bounds__(256)
void k_edge_logits(const int* __restrict__ srcc, const int* __restrict__ dstc,
                   const float* __restrict__ ea_cat, const float* __restrict__ xl,
                   const float* __restrict__ xr, const float* __restrict__ We,
                   const float* __restrict__ att, float* __restrict__ elog,
                   unsigned* __restrict__ mmax, int fout) {
  extern __shared__ float sWe[];                        // EDIM * fout floats
  for (int t = threadIdx.x; t < EDIM * fout; t += blockDim.x) sWe[t] = We[t];
  __syncthreads();

  int lane = threadIdx.x & 31;
  int wave0 = blockIdx.x * (blockDim.x >> 5) + (threadIdx.x >> 5);
  int wstride = gridDim.x * (blockDim.x >> 5);

  for (int e = wave0; e < NETOT; e += wstride) {
    int s = srcc[e], d = dstc[e];
    float eav = ea_cat[(size_t)e * EDIM + lane];        // one edge feat / lane
    float acc = 0.f;
    for (int j = lane; j < fout; j += 32) {
      float v = xl[(size_t)s * fout + j] + xr[(size_t)d * fout + j];
#pragma unroll
      for (int k = 0; k < EDIM; ++k)
        v += __shfl(eav, k, 32) * sWe[k * fout + j];
      v = v > 0.f ? v : NEG_SLOPE * v;
      acc += att[j] * v;
    }
#pragma unroll
    for (int off = 16; off > 0; off >>= 1) acc += __shfl_down(acc, off, 32);
    if (lane == 0) {
      elog[e] = acc;
      atomicMax(&mmax[d], mapf(acc));
    }
  }
}

__global__ void k_edge_exp(const int* __restrict__ dstc, const float* __restrict__ elog,
                           const unsigned* __restrict__ mmax, float* __restrict__ exs,
                           float* __restrict__ den) {
  int i = blockIdx.x * blockDim.x + threadIdx.x;
  if (i >= NETOT) return;
  int d = dstc[i];
  float ex = __expf(elog[i] - unmapf(mmax[d]));
  exs[i] = ex;
  atomicAdd(&den[d], ex);
}

__global__ __launch_bounds__(256)
void k_edge_scatter(const int* __restrict__ srcc, const int* __restrict__ dstc,
                    const float* __restrict__ exs, const float* __restrict__ den,
                    const float* __restrict__ xl, float* __restrict__ out, int fout) {
  int i = blockIdx.x;
  int s = srcc[i], d = dstc[i];
  float a = exs[i] / den[d];
  for (int j = threadIdx.x; j < fout; j += blockDim.x)
    atomicAdd(&out[(size_t)d * fout + j], a * xl[(size_t)s * fout + j]);
}

__global__ void k_bias_relu(float* __restrict__ h, const float* __restrict__ bc,
                            int n, int fout) {
  int i = blockIdx.x * blockDim.x + threadIdx.x;
  if (i < n) h[i] = fmaxf(h[i] + bc[i % fout], 0.f);
}

// ---- head -----------------------------------------------------------------
__global__ void k_pool_cnt(const int* __restrict__ batch, float* __restrict__ cnt) {
  int n = blockIdx.x * blockDim.x + threadIdx.x;
  if (n < NNODES) atomicAdd(&cnt[batch[n]], 1.0f);
}

__global__ __launch_bounds__(256)
void k_pool_sum(const int* __restrict__ batch, const float* __restrict__ h,
                float* __restrict__ pooled) {
  int n = blockIdx.x, j = threadIdx.x;                  // fout == 256
  atomicAdd(&pooled[(size_t)batch[n] * 256 + j], h[(size_t)n * 256 + j]);
}

__global__ void k_fc1(const float* __restrict__ pooled, const float* __restrict__ cnt,
                      const float* __restrict__ w, const float* __restrict__ b,
                      float* __restrict__ z) {
  int t = blockIdx.x * blockDim.x + threadIdx.x;
  if (t >= NGRAPH * 64) return;
  int g = t / 64, c = t % 64;
  float inv = 1.0f / fmaxf(cnt[g], 1.0f);
  float acc = b[c];
  for (int j = 0; j < 256; ++j)
    acc += pooled[(size_t)g * 256 + j] * inv * w[(size_t)j * 64 + c];
  z[t] = acc;
}

__global__ void k_bn_relu(float* __restrict__ z, const float* __restrict__ g,
                          const float* __restrict__ b) {
  int c = threadIdx.x;                                  // 64 threads, 1 block
  float mu = 0.f;
  for (int i = 0; i < NGRAPH; ++i) mu += z[i * 64 + c];
  mu *= (1.0f / NGRAPH);
  float var = 0.f;
  for (int i = 0; i < NGRAPH; ++i) { float d = z[i * 64 + c] - mu; var += d * d; }
  var *= (1.0f / NGRAPH);
  float rs = rsqrtf(var + BN_EPS);
  for (int i = 0; i < NGRAPH; ++i)
    z[i * 64 + c] = fmaxf((z[i * 64 + c] - mu) * rs * g[c] + b[c], 0.f);
}

__global__ void k_fc2(const float* __restrict__ z, const float* __restrict__ w,
                      const float* __restrict__ b, float* __restrict__ out) {
  int gi = blockIdx.x * blockDim.x + threadIdx.x;
  if (gi >= NGRAPH) return;
  float acc = b[0];
  for (int c = 0; c < 64; ++c) acc += z[gi * 64 + c] * w[c];
  out[gi] = acc;
}

// ---------------------------------------------------------------------------
static inline char* carve(char*& p, size_t bytes) {
  char* r = p;
  p += (bytes + 255) & ~(size_t)255;
  return r;
}
static inline int cdiv(long a, long b) { return (int)((a + b - 1) / b); }

extern "C" void kernel_launch(void* const* d_in, const int* in_sizes, int n_in,
                              void* d_out, int out_size, void* d_ws, size_t ws_size,
                              hipStream_t stream) {
  const float* x     = (const float*)d_in[0];
  const int*   ei    = (const int*)  d_in[1];
  const float* eattr = (const float*)d_in[2];
  const int*   batch = (const int*)  d_in[3];
  const float *Wl[3], *bl[3], *Wr[3], *br[3], *We[3], *att[3], *bc[3];
  int base = 4;
  for (int i = 0; i < 3; ++i) {
    Wl[i]  = (const float*)d_in[base + 0];
    bl[i]  = (const float*)d_in[base + 1];
    Wr[i]  = (const float*)d_in[base + 2];
    br[i]  = (const float*)d_in[base + 3];
    We[i]  = (const float*)d_in[base + 4];
    att[i] = (const float*)d_in[base + 5];
    bc[i]  = (const float*)d_in[base + 6];
    base += 7;
  }
  const float* fc1_w = (const float*)d_in[25];
  const float* fc1_b = (const float*)d_in[26];
  const float* bn_g  = (const float*)d_in[27];
  const float* bn_b  = (const float*)d_in[28];
  const float* fc2_w = (const float*)d_in[29];
  const float* fc2_b = (const float*)d_in[30];

  // ---- workspace carve (~315 MB total) ----
  char* p = (char*)d_ws;
  float*  bufA   = (float*) carve(p, (size_t)NNODES * 1024 * 4);
  float*  bufB   = (float*) carve(p, (size_t)NNODES * 1024 * 4);
  __bf16* hbf    = (__bf16*)carve(p, (size_t)NNODES * 1024 * 2);
  __bf16* wbf_l  = (__bf16*)carve(p, (size_t)1024 * 1024 * 2);
  __bf16* wbf_r  = (__bf16*)carve(p, (size_t)1024 * 1024 * 2);
  float*  xl     = (float*) carve(p, (size_t)NNODES * 1024 * 4);
  float*  xr     = (float*) carve(p, (size_t)NNODES * 1024 * 4);
  float*  ea_cat = (float*) carve(p, (size_t)NETOT * EDIM * 4);
  int*    srcc   = (int*)   carve(p, (size_t)NETOT * 4);
  int*    dstc   = (int*)   carve(p, (size_t)NETOT * 4);
  float*  deg    = (float*) carve(p, (size_t)NNODES * 4);
  float*  elog   = (float*) carve(p, (size_t)NETOT * 4);
  float*  exs    = (float*) carve(p, (size_t)NETOT * 4);
  unsigned* mmax = (unsigned*)carve(p, (size_t)NNODES * 4);
  float*  den    = (float*) carve(p, (size_t)NNODES * 4);
  float*  cnt    = (float*) carve(p, (size_t)NGRAPH * 4);
  float*  pooled = (float*) carve(p, (size_t)NGRAPH * 256 * 4);
  float*  zbuf   = (float*) carve(p, (size_t)NGRAPH * 64 * 4);
  (void)ws_size; (void)n_in; (void)in_sizes; (void)out_size;

  const int B = 256;

  // ---- graph setup (layer independent) ----
  hipMemsetAsync(deg, 0, (size_t)NNODES * 4, stream);
  hipMemsetAsync(ea_cat + (size_t)NEDGES * EDIM, 0, (size_t)NNODES * EDIM * 4, stream);
  hipMemcpyAsync(ea_cat, eattr, (size_t)NEDGES * EDIM * 4, hipMemcpyDeviceToDevice, stream);
  k_fill_cat<<<cdiv(NETOT, B), B, 0, stream>>>(ei, srcc, dstc);
  k_deg<<<cdiv(NEDGES, B), B, 0, stream>>>(ei, deg);
  k_loop_acc<<<cdiv((long)NEDGES * EDIM, B), B, 0, stream>>>(ei, eattr, ea_cat);
  k_loop_div<<<cdiv((long)NNODES * EDIM, B), B, 0, stream>>>(deg, ea_cat);

  // ---- three GATv2 layers ----
  const int fin_a[3]  = {256, 512, 1024};
  const int fout_a[3] = {512, 1024, 256};
  const float* hin = x;
  float* hout[3] = {bufA, bufB, bufA};

  for (int l = 0; l < 3; ++l) {
    int fin = fin_a[l], fout = fout_a[l];
    long nh = (long)NNODES * fin;
    long npack = ((long)fout / 16) * (fin / 32) * 32;   // repack threads

    k_f2bf<<<cdiv(nh, B), B, 0, stream>>>(hin, hbf, (int)nh);
    k_repackW<<<cdiv(npack, B), B, 0, stream>>>(Wl[l], wbf_l, fin, fout);
    k_repackW<<<cdiv(npack, B), B, 0, stream>>>(Wr[l], wbf_r, fin, fout);

    long waves = (long)(NNODES / 16) * (fout / 64);
    int gblocks = cdiv(waves * 32, B);
    k_gemm_bf16<<<gblocks, B, 0, stream>>>(hbf, wbf_l, bl[l], xl, NNODES, fin, fout);
    k_gemm_bf16<<<gblocks, B, 0, stream>>>(hbf, wbf_r, br[l], xr, NNODES, fin, fout);

    hipMemsetAsync(mmax, 0, (size_t)NNODES * 4, stream);       // mapped -inf
    hipMemsetAsync(den,  0, (size_t)NNODES * 4, stream);
    hipMemsetAsync(hout[l], 0, (size_t)NNODES * fout * 4, stream);

    size_t shmem = (size_t)EDIM * fout * 4;                    // <= 128 KB
    k_edge_logits<<<2048, B, shmem, stream>>>(
        srcc, dstc, ea_cat, xl, xr, We[l], att[l], elog, mmax, fout);
    k_edge_exp<<<cdiv(NETOT, B), B, 0, stream>>>(dstc, elog, mmax, exs, den);
    k_edge_scatter<<<NETOT, B, 0, stream>>>(srcc, dstc, exs, den, xl, hout[l], fout);
    k_bias_relu<<<cdiv((long)NNODES * fout, B), B, 0, stream>>>(
        hout[l], bc[l], NNODES * fout, fout);

    hin = hout[l];
  }

  // ---- mean pool + MLP head ----
  hipMemsetAsync(cnt, 0, (size_t)NGRAPH * 4, stream);
  hipMemsetAsync(pooled, 0, (size_t)NGRAPH * 256 * 4, stream);
  k_pool_cnt<<<cdiv(NNODES, B), B, 0, stream>>>(batch, cnt);
  k_pool_sum<<<NNODES, 256, 0, stream>>>(batch, hout[2], pooled);
  k_fc1<<<cdiv(NGRAPH * 64, B), B, 0, stream>>>(pooled, cnt, fc1_w, fc1_b, zbuf);
  k_bn_relu<<<1, 64, 0, stream>>>(zbuf, bn_g, bn_b);
  k_fc2<<<1, NGRAPH, 0, stream>>>(zbuf, fc2_w, fc2_b, (float*)d_out);
}
